// FeedForwardNetwork_73358041415812
// MI455X (gfx1250) — 8-bit matrix operand order repaired and checked by reading, compiled, not run on hardware
//
#include <hip/hip_runtime.h>
#include <cstdint>

typedef int v8i __attribute__((ext_vector_type(8)));

typedef int v4i_vs __attribute__((__vector_size__(16)));
typedef __attribute__((address_space(1))) v4i_vs glob_v4;
typedef __attribute__((address_space(3))) v4i_vs lds_v4;

#if __has_builtin(__builtin_amdgcn_global_load_async_to_lds_b128)
#define HAVE_ASYNC_LDS 1
#else
#define HAVE_ASYNC_LDS 0
#endif

#if __has_builtin(__builtin_amdgcn_s_wait_asynccnt)
#define WAIT_ASYNC0() __builtin_amdgcn_s_wait_asynccnt(0)
#else
#define WAIT_ASYNC0() asm volatile("s_wait_asynccnt 0x0" ::: "memory")
#endif

#define EPSQ 1e-5f

static constexpr int TOK = 8192;
static constexpr int DM  = 2048;
static constexpr int DH  = 8192;
static constexpr long NW = (long)DH * DM;

static constexpr size_t OFF_W1Q = 0;
static constexpr size_t OFF_W2Q = OFF_W1Q + (size_t)NW;
static constexpr size_t OFF_XQ  = OFF_W2Q + (size_t)NW;
static constexpr size_t OFF_HQ  = OFF_XQ  + (size_t)TOK*DM;
static constexpr size_t OFF_H   = OFF_HQ  + (size_t)TOK*DH;
static constexpr size_t OFF_SX  = OFF_H   + (size_t)TOK*DH*4;
static constexpr size_t OFF_SH  = OFF_SX  + (size_t)TOK*4;
static constexpr size_t OFF_SW1 = OFF_SH  + (size_t)TOK*4;
static constexpr size_t OFF_SW2 = OFF_SW1 + 256;
static constexpr size_t OFF_P1  = OFF_SW2 + 256;
static constexpr size_t OFF_P2  = OFF_P1  + 4096*4;

__global__ __launch_bounds__(256) void absSumPartial(const float* __restrict__ w,
                                                     float* __restrict__ partial,
                                                     int nPerBlock, long nTotal) {
  __shared__ float s[256];
  const int t = threadIdx.x;
  const long base = (long)blockIdx.x * nPerBlock;
  float acc = 0.f;
  for (int i = t; i < nPerBlock; i += 256) {
    long idx = base + i;
    if (idx < nTotal) acc += fabsf(w[idx]);
  }
  s[t] = acc;
  __syncthreads();
  for (int off = 128; off > 0; off >>= 1) {
    if (t < off) s[t] += s[t + off];
    __syncthreads();
  }
  if (t == 0) partial[blockIdx.x] = s[0];
}

__global__ __launch_bounds__(256) void absSumFinal(const float* __restrict__ partial,
                                                   int np, float invN,
                                                   float* __restrict__ scalePair) {
  __shared__ float s[256];
  const int t = threadIdx.x;
  float acc = 0.f;
  for (int i = t; i < np; i += 256) acc += partial[i];
  s[t] = acc;
  __syncthreads();
  for (int off = 128; off > 0; off >>= 1) {
    if (t < off) s[t] += s[t + off];
    __syncthreads();
  }
  if (t == 0) {
    float mean = s[0] * invN;
    float dq = fmaxf(mean, EPSQ);
    scalePair[0] = dq;
    scalePair[1] = 1.0f / dq;
  }
}

__global__ __launch_bounds__(256) void quantWeight(const float* __restrict__ w,
                                                   const float* __restrict__ scalePair,
                                                   int8_t* __restrict__ q, long n) {
  const float scale = scalePair[1];
  long i = (long)blockIdx.x * 256 + threadIdx.x;
  const long stride = (long)gridDim.x * 256;
  for (; i < n; i += stride) {
    float v = rintf(w[i] * scale);
    v = fminf(fmaxf(v, -1.f), 1.f);
    q[i] = (int8_t)v;
  }
}

__global__ __launch_bounds__(256) void quantAct(const float* __restrict__ x,
                                                int8_t* __restrict__ q,
                                                float* __restrict__ rowdq, int W) {
  __shared__ float s[256];
  const long row = blockIdx.x;
  const float* xr = x + row * (long)W;
  int8_t* qr = q + row * (long)W;
  const int t = threadIdx.x;
  float amax = 0.f;
  for (int i = t; i < W; i += 256) amax = fmaxf(amax, fabsf(xr[i]));
  s[t] = amax;
  __syncthreads();
  for (int off = 128; off > 0; off >>= 1) {
    if (t < off) s[t] = fmaxf(s[t], s[t + off]);
    __syncthreads();
  }
  const float am = fmaxf(s[0], EPSQ);
  const float scale = 127.0f / am;
  for (int i = t; i < W; i += 256) {
    float v = rintf(xr[i] * scale);
    v = fminf(fmaxf(v, -128.f), 127.f);
    qr[i] = (int8_t)v;
  }
  if (t == 0) rowdq[row] = am / 127.0f;
}

template <bool GELU>
__global__ __launch_bounds__(256) __attribute__((amdgpu_waves_per_eu(2)))
void gemm_i8_wmma(const int8_t* __restrict__ A,
                  const int8_t* __restrict__ B,
                  const float* __restrict__ rowdq,
                  const float* __restrict__ wPair,
                  float* __restrict__ C,
                  int N, int K) {
  __shared__ uint32_t aT[2][64 * 32];
  __shared__ uint32_t bT[2][128 * 32];
  const int t    = threadIdx.x;
  const int lane = t & 31;
  const int wid  = t >> 5;
  const int half = lane >> 4;
  const int lr   = lane & 15;
  const int m0 = (wid & 1) * 32;
  const int n0 = (wid >> 1) * 32;
  const long row0 = (long)blockIdx.y * 64;
  const long col0 = (long)blockIdx.x * 128;
  const float wdq = wPair[0];

  const int nk = K >> 7;

  v8i c00 = {}, c01 = {}, c10 = {}, c11 = {};

#if HAVE_ASYNC_LDS
#pragma unroll
  for (int j = 0; j < 2; ++j) {
    const int idx = t + j * 256, row = idx >> 3, seg = idx & 7;
    __builtin_amdgcn_global_load_async_to_lds_b128(
        (glob_v4*)(A + (row0 + row) * (long)K + seg * 16),
        (lds_v4*)((char*)&aT[0][0] + idx * 16), 0, 0);
  }
#pragma unroll
  for (int j = 0; j < 4; ++j) {
    const int idx = t + j * 256, row = idx >> 3, seg = idx & 7;
    __builtin_amdgcn_global_load_async_to_lds_b128(
        (glob_v4*)(B + (col0 + row) * (long)K + seg * 16),
        (lds_v4*)((char*)&bT[0][0] + idx * 16), 0, 0);
  }
  WAIT_ASYNC0();
  __syncthreads();

  for (int ki = 0; ki < nk; ++ki) {
    const int cur = ki & 1;
    const bool more = (ki + 1) < nk;

    if (more) {
      const long kb = (long)(ki + 1) << 7;
#pragma unroll
      for (int j = 0; j < 2; ++j) {
        const int idx = t + j * 256, row = idx >> 3, seg = idx & 7;
        __builtin_amdgcn_global_load_async_to_lds_b128(
            (glob_v4*)(A + (row0 + row) * (long)K + kb + seg * 16),
            (lds_v4*)((char*)&aT[cur ^ 1][0] + idx * 16), 0, 0);
      }
#pragma unroll
      for (int j = 0; j < 4; ++j) {
        const int idx = t + j * 256, row = idx >> 3, seg = idx & 7;
        __builtin_amdgcn_global_load_async_to_lds_b128(
            (glob_v4*)(B + (col0 + row) * (long)K + kb + seg * 16),
            (lds_v4*)((char*)&bT[cur ^ 1][0] + idx * 16), 0, 0);
      }
    }

    const uint32_t* As = aT[cur];
    const uint32_t* Bs = bT[cur];
#pragma unroll
    for (int s = 0; s < 2; ++s) {
      v8i a0, a1, b0, b1;
#pragma unroll
      for (int v = 0; v < 8; ++v) {
        const int ka  = (v >> 1) * 4 + (v & 1) + half * 2;
        const int kbw = (v >> 1) * 4 + (v & 1) + half * 2;
        a0[v] = (int)As[(m0 + lr)      * 32 + s * 16 + ka];
        a1[v] = (int)As[(m0 + 16 + lr) * 32 + s * 16 + ka];
        b0[v] = (int)Bs[(n0 + lr)      * 32 + s * 16 + kbw];
        b1[v] = (int)Bs[(n0 + 16 + lr) * 32 + s * 16 + kbw];
      }
      c00 = __builtin_amdgcn_wmma_i32_16x16x64_iu8(true, a0, true, b0, c00, false, false);
      c01 = __builtin_amdgcn_wmma_i32_16x16x64_iu8(true, a0, true, b1, c01, false, false);
      c10 = __builtin_amdgcn_wmma_i32_16x16x64_iu8(true, a1, true, b0, c10, false, false);
      c11 = __builtin_amdgcn_wmma_i32_16x16x64_iu8(true, a1, true, b1, c11, false, false);
    }

    WAIT_ASYNC0();
    __syncthreads();
  }

#else
  const int K16 = K >> 4;
  const int4* gA = (const int4*)A;
  const int4* gB = (const int4*)B;
#pragma unroll
  for (int j = 0; j < 2; ++j) {
    const int idx = t + j * 256, row = idx >> 3, seg = idx & 7;
    ((int4*)aT[0])[idx] = gA[(row0 + row) * K16 + seg];
  }
#pragma unroll
  for (int j = 0; j < 4; ++j) {
    const int idx = t + j * 256, row = idx >> 3, seg = idx & 7;
    ((int4*)bT[0])[idx] = gB[(col0 + row) * K16 + seg];
  }
  __syncthreads();

  for (int ki = 0; ki < nk; ++ki) {
    const int cur = ki & 1;
    const bool more = (ki + 1) < nk;

    int4 na[2], nb[4];
    if (more) {
      const int kseg = ((ki + 1) << 7) >> 4;
#pragma unroll
      for (int j = 0; j < 2; ++j) {
        const int idx = t + j * 256, row = idx >> 3, seg = idx & 7;
        na[j] = gA[(row0 + row) * K16 + kseg + seg];
      }
#pragma unroll
      for (int j = 0; j < 4; ++j) {
        const int idx = t + j * 256, row = idx >> 3, seg = idx & 7;
        nb[j] = gB[(col0 + row) * K16 + kseg + seg];
      }
    }

    const uint32_t* As = aT[cur];
    const uint32_t* Bs = bT[cur];
#pragma unroll
    for (int s = 0; s < 2; ++s) {
      v8i a0, a1, b0, b1;
#pragma unroll
      for (int v = 0; v < 8; ++v) {
        const int ka  = (v >> 1) * 4 + (v & 1) + half * 2;
        const int kbw = (v >> 1) * 4 + (v & 1) + half * 2;
        a0[v] = (int)As[(m0 + lr)      * 32 + s * 16 + ka];
        a1[v] = (int)As[(m0 + 16 + lr) * 32 + s * 16 + ka];
        b0[v] = (int)Bs[(n0 + lr)      * 32 + s * 16 + kbw];
        b1[v] = (int)Bs[(n0 + 16 + lr) * 32 + s * 16 + kbw];
      }
      c00 = __builtin_amdgcn_wmma_i32_16x16x64_iu8(true, a0, true, b0, c00, false, false);
      c01 = __builtin_amdgcn_wmma_i32_16x16x64_iu8(true, a0, true, b1, c01, false, false);
      c10 = __builtin_amdgcn_wmma_i32_16x16x64_iu8(true, a1, true, b0, c10, false, false);
      c11 = __builtin_amdgcn_wmma_i32_16x16x64_iu8(true, a1, true, b1, c11, false, false);
    }

    if (more) {
#pragma unroll
      for (int j = 0; j < 2; ++j) ((int4*)aT[cur ^ 1])[t + j * 256] = na[j];
#pragma unroll
      for (int j = 0; j < 4; ++j) ((int4*)bT[cur ^ 1])[t + j * 256] = nb[j];
    }
    __syncthreads();
  }
#endif

#pragma unroll
  for (int r = 0; r < 8; ++r) {
    const long gm0 = row0 + m0 + r + half * 8;
    const long gm1 = gm0 + 16;
    const float dq0 = rowdq[gm0] * wdq;
    const float dq1 = rowdq[gm1] * wdq;
    const long gn0 = col0 + n0 + lr;
    const long gn1 = gn0 + 16;
    float v00 = (float)c00[r] * dq0;
    float v01 = (float)c01[r] * dq0;
    float v10 = (float)c10[r] * dq1;
    float v11 = (float)c11[r] * dq1;
    if (GELU) {
      v00 = 0.5f * v00 * (1.0f + erff(v00 * 0.70710678118654752f));
      v01 = 0.5f * v01 * (1.0f + erff(v01 * 0.70710678118654752f));
      v10 = 0.5f * v10 * (1.0f + erff(v10 * 0.70710678118654752f));
      v11 = 0.5f * v11 * (1.0f + erff(v11 * 0.70710678118654752f));
    }
    C[gm0 * (long)N + gn0] = v00;
    C[gm0 * (long)N + gn1] = v01;
    C[gm1 * (long)N + gn0] = v10;
    C[gm1 * (long)N + gn1] = v11;
  }
}

extern "C" void kernel_launch(void* const* d_in, const int* in_sizes, int n_in,
                              void* d_out, int out_size, void* d_ws, size_t ws_size,
                              hipStream_t stream) {
  (void)in_sizes; (void)n_in; (void)out_size; (void)ws_size;
  const float* x  = (const float*)d_in[0];
  const float* w1 = (const float*)d_in[1];
  const float* w2 = (const float*)d_in[2];
  float* out = (float*)d_out;
  char* ws = (char*)d_ws;

  int8_t* w1q = (int8_t*)(ws + OFF_W1Q);
  int8_t* w2q = (int8_t*)(ws + OFF_W2Q);
  int8_t* xq  = (int8_t*)(ws + OFF_XQ);
  int8_t* hq  = (int8_t*)(ws + OFF_HQ);
  float*  h   = (float*)(ws + OFF_H);
  float*  sx  = (float*)(ws + OFF_SX);
  float*  sh  = (float*)(ws + OFF_SH);
  float*  sw1 = (float*)(ws + OFF_SW1);
  float*  sw2 = (float*)(ws + OFF_SW2);
  float*  p1  = (float*)(ws + OFF_P1);
  float*  p2  = (float*)(ws + OFF_P2);

  absSumPartial<<<4096, 256, 0, stream>>>(w1, p1, 4096, NW);
  absSumFinal<<<1, 256, 0, stream>>>(p1, 4096, 1.0f / (float)NW, sw1);
  absSumPartial<<<4096, 256, 0, stream>>>(w2, p2, 4096, NW);
  absSumFinal<<<1, 256, 0, stream>>>(p2, 4096, 1.0f / (float)NW, sw2);

  quantWeight<<<8192, 256, 0, stream>>>(w1, sw1, w1q, NW);
  quantWeight<<<8192, 256, 0, stream>>>(w2, sw2, w2q, NW);

  quantAct<<<TOK, 256, 0, stream>>>(x, xq, sx, DM);

  dim3 g1(DH / 128, TOK / 64);
  gemm_i8_wmma<true><<<g1, 256, 0, stream>>>(xq, w1q, sx, sw1, h, DH, DM);

  quantAct<<<TOK, 256, 0, stream>>>(h, hq, sh, DH);

  dim3 g2(DM / 128, TOK / 64);
  gemm_i8_wmma<false><<<g2, 256, 0, stream>>>(hq, w2q, sh, sw2, out, DM, DH);
}
